// MultiheadAttention_20650202759194
// MI455X (gfx1250) — compile-verified, matrix operand order checked by reading, not run on hardware
//
#include <hip/hip_runtime.h>
#include <hip/hip_bf16.h>

// MI455X / gfx1250 multi-head attention forward.
// d_out = [ out (8*1024*1024 f32) | attn (8*16*1024*1024 f32) ]
// d_ws  = q,k,v,x scratch held in bf16: 4 * 8192*1024 * 2B = 64 MB.

typedef __attribute__((ext_vector_type(16))) __bf16 v16bf;
typedef __attribute__((ext_vector_type(8)))  float  v8f;

#define WMMA_BF16(a, b, c) \
  __builtin_amdgcn_wmma_f32_16x16x32_bf16(false, (a), false, (b), (short)0, (c), false, false)

// Async 16-byte global->LDS copy (gfx1250 GLOBAL_LOAD_ASYNC_TO_LDS_B128,
// tracked by ASYNCcnt). LDS operand is the addrspace(3) byte offset, which is
// the low 32 bits of the generic pointer (ISA 10.2).
__device__ __forceinline__ void async_copy16(const void* gptr, void* lptr) {
  unsigned lds = (unsigned)(uintptr_t)lptr;
  asm volatile("global_load_async_to_lds_b128 %0, %1, off"
               :: "v"(lds), "v"(gptr) : "memory");
}
#define WAIT_ASYNC(n) asm volatile("s_wait_asynccnt %0" :: "i"(n) : "memory")

// K-index inside a 16x32 bf16 A-frag (and mirrored 32x16 B-frag) for element
// e (0..15), given lane-half. Per CDNA5 ISA §7.12.2: lanes 0-15
// hold K=0..7,16..23; lanes 16-31 hold K=8..15,24..31 (pairs per VGPR).
// This maps to two contiguous 16B LDS reads per lane -> ds_load_b128 pairs.
__device__ __forceinline__ int frag_k(int e, int hf) {
  return (e & 1) + 2 * ((e >> 1) & 3) + 16 * (e >> 3) + 8 * hf;
}

// ---------------------------------------------------------------------------
// C[M,Nout] = A[M,K] @ W[Nout,K]^T + bias   (torch Linear semantics)
// TA: float (first use of activations) or __bf16 (pre-quantized scratch).
// TC: __bf16 (internal scratch) or float (final output).
// Block tile 128x64, K-step 32, 8 waves; wave w owns rows [16w,16w+16).
// ---------------------------------------------------------------------------
template <typename TA, typename TC>
__global__ __launch_bounds__(256)
void gemm_bias_kernel(const TA* __restrict__ A, const float* __restrict__ W,
                      const float* __restrict__ bias, TC* __restrict__ C,
                      int M, int Nout, int K) {
  const int PAD = 40;                       // bf16 row stride (32 + 8 pad)
  __shared__ __align__(16) __bf16 sA[128 * PAD];
  __shared__ __align__(16) __bf16 sW[64 * PAD];

  const int tid = threadIdx.x;
  const int wave = tid >> 5, lane = tid & 31;
  const int hf = lane >> 4, l16 = lane & 15;
  const int m0 = blockIdx.x * 128;
  const int n0 = blockIdx.y * 64;

  v8f acc[4] = {};

  for (int k0 = 0; k0 < K; k0 += 32) {
    __syncthreads();
    // Stage A tile 128x32 -> bf16 LDS (coalesced; cvt only if TA==float)
#pragma unroll
    for (int i = 0; i < 4; ++i) {
      int idx = tid + i * 256;
      int r = idx >> 3, c4 = (idx & 7) * 4;
      __bf16* d = &sA[r * PAD + c4];
      if constexpr (sizeof(TA) == 4) {
        float4 f = *(const float4*)(A + (size_t)(m0 + r) * K + k0 + c4);
        d[0] = (__bf16)f.x; d[1] = (__bf16)f.y; d[2] = (__bf16)f.z; d[3] = (__bf16)f.w;
      } else {
        *(uint2*)d = *(const uint2*)(A + (size_t)(m0 + r) * K + k0 + c4);
      }
    }
    // Stage W tile 64x32 (row-major along K == B-frag orientation)
#pragma unroll
    for (int i = 0; i < 2; ++i) {
      int idx = tid + i * 256;
      int r = idx >> 3, c4 = (idx & 7) * 4;
      float4 f = *(const float4*)(W + (size_t)(n0 + r) * K + k0 + c4);
      __bf16* d = &sW[r * PAD + c4];
      d[0] = (__bf16)f.x; d[1] = (__bf16)f.y; d[2] = (__bf16)f.z; d[3] = (__bf16)f.w;
    }
    if (k0 + 32 < K) {                      // hint next K-step into cache
      __builtin_prefetch(A + (size_t)(m0 + (tid >> 3)) * K + k0 + 32, 0, 1);
      __builtin_prefetch(W + (size_t)(n0 + (tid & 63)) * K + k0 + 32, 0, 1);
    }
    __syncthreads();

    v16bf a;
#pragma unroll
    for (int e = 0; e < 16; ++e)
      a[e] = sA[(wave * 16 + l16) * PAD + frag_k(e, hf)];
#pragma unroll
    for (int ns = 0; ns < 4; ++ns) {
      v16bf b;
#pragma unroll
      for (int e = 0; e < 16; ++e)
        b[e] = sW[(ns * 16 + l16) * PAD + frag_k(e, hf)];
      acc[ns] = WMMA_BF16(a, b, acc[ns]);
    }
  }

  // Epilogue: C/D layout -> row = r + 8*(lane/16), col = lane%16
#pragma unroll
  for (int ns = 0; ns < 4; ++ns) {
    int col = n0 + ns * 16 + l16;
    float bv = bias[col];
#pragma unroll
    for (int r = 0; r < 8; ++r) {
      int row = m0 + wave * 16 + hf * 8 + r;
      float val = acc[ns][r] + bv;
      if constexpr (sizeof(TC) == 4) C[(size_t)row * Nout + col] = val;
      else                           C[(size_t)row * Nout + col] = (__bf16)val;
    }
  }
}

// ---------------------------------------------------------------------------
// Fused attention: one block per (b, h, 32-query-row block).
// All 1024 key-columns of scores in LDS (exact softmax, attn hits HBM once).
// K/V tiles double-buffered via GLOBAL_LOAD_ASYNC_TO_LDS_B128 (ASYNCcnt),
// overlapping HBM/L2 latency with WMMA. ~152 KB LDS.
// ---------------------------------------------------------------------------
__global__ __launch_bounds__(256)
void attn_kernel(const __bf16* __restrict__ Q, const __bf16* __restrict__ Km,
                 const __bf16* __restrict__ V, float* __restrict__ attn_out,
                 __bf16* __restrict__ X) {
  const int N = 1024, D = 1024, DH = 64, H = 16;
  const int SP = 1028;                      // score row stride (mod 64 != 0)
  __shared__ __align__(16) float  sS[32 * SP];      // 128.5 KB scores/probs
  __shared__ __align__(16) __bf16 sQ[32 * 72];      //   4.5 KB
  __shared__ __align__(16) __bf16 sKV[2][64 * 72];  //  18.0 KB double buffer
  __shared__ float sred[32 * 8];                    //   1.0 KB

  const int tid = threadIdx.x;
  const int wave = tid >> 5, lane = tid & 31;
  const int hf = lane >> 4, l16 = lane & 15;
  const int msub = wave & 1, nsub = wave >> 1;   // 8 waves -> 2x4 16x16 subtiles
  const int rb = blockIdx.x, h = blockIdx.y, b = blockIdx.z;

  const size_t qbase  = ((size_t)b * N + rb * 32) * D + h * DH;
  const size_t kvbase = ((size_t)b * N) * D + h * DH;

  const int cr = tid >> 3, cc8 = (tid & 7) * 8;  // 16B-chunk coords (8 chunks/row)

  // Async-stage Q tile (32x64, 1 chunk/thread) and K tile 0 (64x64, 2/thread).
  async_copy16(Q + qbase + (size_t)cr * D + cc8, &sQ[cr * 72 + cc8]);
#pragma unroll
  for (int i = 0; i < 2; ++i) {
    int idx = tid + i * 256, r = idx >> 3, c8 = (idx & 7) * 8;
    async_copy16(Km + kvbase + (size_t)r * D + c8, &sKV[0][r * 72 + c8]);
  }

  // ---- Pass 1: S = (Q K^T) * scale, all 1024 columns into LDS ----
  const float scale = 0.125f;               // Dh^-0.5
  for (int j = 0; j < 16; ++j) {
    const int cur = j & 1;
    if (j + 1 < 16) {                       // issue next tile, then wait for cur
#pragma unroll
      for (int i = 0; i < 2; ++i) {
        int idx = tid + i * 256, r = idx >> 3, c8 = (idx & 7) * 8;
        async_copy16(Km + kvbase + (size_t)((j + 1) * 64 + r) * D + c8,
                     &sKV[cur ^ 1][r * 72 + c8]);
      }
      WAIT_ASYNC(2);
    } else {
      WAIT_ASYNC(0);
    }
    __syncthreads();                        // tile j visible to all waves

    v8f acc = {};
#pragma unroll
    for (int ks = 0; ks < 2; ++ks) {        // contraction over Dh=64
      v16bf a, bb;
#pragma unroll
      for (int e = 0; e < 16; ++e)
        a[e] = sQ[(msub * 16 + l16) * 72 + ks * 32 + frag_k(e, hf)];
#pragma unroll
      for (int e = 0; e < 16; ++e)
        bb[e] = sKV[cur][(nsub * 16 + l16) * 72 + ks * 32 + frag_k(e, hf)];
      acc = WMMA_BF16(a, bb, acc);
    }
#pragma unroll
    for (int r = 0; r < 8; ++r)
      sS[(msub * 16 + hf * 8 + r) * SP + j * 64 + nsub * 16 + l16] = acc[r] * scale;
    __syncthreads();                        // done reading cur before overwrite
  }

  // ---- Exact softmax over 1024 cols; 8 threads per row ----
  const int row = tid >> 3, sub = tid & 7;
  float m = -3.4e38f;
  for (int c = sub; c < N; c += 8) m = fmaxf(m, sS[row * SP + c]);
  sred[row * 8 + sub] = m;
  __syncthreads();
#pragma unroll
  for (int i = 0; i < 8; ++i) m = fmaxf(m, sred[row * 8 + i]);
  float s = 0.f;
  for (int c = sub; c < N; c += 8) {
    float e = __expf(sS[row * SP + c] - m);
    sS[row * SP + c] = e;
    s += e;
  }
  __syncthreads();
  sred[row * 8 + sub] = s;
  __syncthreads();
  s = 0.f;
#pragma unroll
  for (int i = 0; i < 8; ++i) s += sred[row * 8 + i];
  const float inv = 1.f / s;

  // Normalize in LDS; write attn to HBM exactly once.
  const size_t abase = (((size_t)(b * H + h)) * N + rb * 32 + row) * N;
  for (int c = sub; c < N; c += 8) {
    float p = sS[row * SP + c] * inv;
    sS[row * SP + c] = p;
    attn_out[abase + c] = p;
  }
  __syncthreads();

  // ---- Pass 2: X = P @ V (probs from LDS w/ on-the-fly bf16 cvt) ----
#pragma unroll
  for (int i = 0; i < 2; ++i) {             // async-stage V tile 0
    int idx = tid + i * 256, r = idx >> 3, c8 = (idx & 7) * 8;
    async_copy16(V + kvbase + (size_t)r * D + c8, &sKV[0][r * 72 + c8]);
  }
  v8f acc = {};
  for (int j = 0; j < 16; ++j) {
    const int cur = j & 1;
    if (j + 1 < 16) {
#pragma unroll
      for (int i = 0; i < 2; ++i) {
        int idx = tid + i * 256, r = idx >> 3, c8 = (idx & 7) * 8;
        async_copy16(V + kvbase + (size_t)((j + 1) * 64 + r) * D + c8,
                     &sKV[cur ^ 1][r * 72 + c8]);
      }
      WAIT_ASYNC(2);
    } else {
      WAIT_ASYNC(0);
    }
    __syncthreads();
#pragma unroll
    for (int ks = 0; ks < 2; ++ks) {
      v16bf a, bb;
#pragma unroll
      for (int e = 0; e < 16; ++e)
        a[e] = (__bf16)sS[(msub * 16 + l16) * SP + j * 64 + ks * 32 + frag_k(e, hf)];
#pragma unroll
      for (int e = 0; e < 16; ++e)          // B[k][n] = V[key][dh]
        bb[e] = sKV[cur][(ks * 32 + frag_k(e, hf)) * 72 + nsub * 16 + l16];
      acc = WMMA_BF16(a, bb, acc);
    }
    __syncthreads();
  }
#pragma unroll
  for (int r = 0; r < 8; ++r) {
    size_t grow = (size_t)b * N + rb * 32 + msub * 16 + hf * 8 + r;
    X[grow * D + h * DH + nsub * 16 + l16] = (__bf16)acc[r];
  }
}

// ---------------------------------------------------------------------------
extern "C" void kernel_launch(void* const* d_in, const int* in_sizes, int n_in,
                              void* d_out, int out_size, void* d_ws, size_t ws_size,
                              hipStream_t stream) {
  const float* x_q  = (const float*)d_in[0];
  const float* x_kv = (const float*)d_in[1];
  const float* Wq = (const float*)d_in[2]; const float* bq = (const float*)d_in[3];
  const float* Wk = (const float*)d_in[4]; const float* bk = (const float*)d_in[5];
  const float* Wv = (const float*)d_in[6]; const float* bv = (const float*)d_in[7];
  const float* Wo = (const float*)d_in[8]; const float* bo = (const float*)d_in[9];

  const int B = 8, N = 1024, D = 1024;
  const size_t BN = (size_t)B * N;          // 8192 rows
  const size_t mat = BN * D;                // 8,388,608 elems

  float* out  = (float*)d_out;
  float* attn = out + mat;                  // tuple: (out, attn)

  __bf16* q = (__bf16*)d_ws;                // bf16 scratch: 64 MB total
  __bf16* k = q + mat;
  __bf16* v = k + mat;
  __bf16* x = v + mat;

  dim3 blk(256);
  dim3 gridG(BN / 128, D / 64);             // 64 x 16

  gemm_bias_kernel<float, __bf16><<<gridG, blk, 0, stream>>>(x_q,  Wq, bq, q, (int)BN, D, D);
  gemm_bias_kernel<float, __bf16><<<gridG, blk, 0, stream>>>(x_kv, Wk, bk, k, (int)BN, D, D);
  gemm_bias_kernel<float, __bf16><<<gridG, blk, 0, stream>>>(x_kv, Wv, bv, v, (int)BN, D, D);

  dim3 gridA(N / 32, 16, B);                // (row-blocks, H, B)
  attn_kernel<<<gridA, blk, 0, stream>>>(q, k, v, attn, x);

  gemm_bias_kernel<__bf16, float><<<gridG, blk, 0, stream>>>(x, Wo, bo, out, (int)BN, D, D);
}
